// Attention_14542759264705
// MI455X (gfx1250) — compile-verified, matrix operand order checked by reading, not run on hardware
//
#include <hip/hip_runtime.h>
#include <hip/hip_bf16.h>

// Attention block for MI455X (gfx1250), wave32, WMMA f16 (f32 accumulate).
// Pipeline: f32->f16 convert, Q/K/V projections (async-LDS double-buffered
// WMMA GEMM, V stored transposed), RoPE, flash-attention (WMMA scores + PV,
// online softmax), output projection (same GEMM, fp32 store).
// Compute-bound: ~155 GFLOP vs ~120 MB resident (fits in 192 MB L2).
// Requires ~80 MB workspace.

#define S_LEN 2048
#define NH    32
#define NKV   8
#define HD    64
#define DIMV  2048
#define MTOT  4096   // B * S

typedef __attribute__((ext_vector_type(16))) _Float16 v16h;
typedef __attribute__((ext_vector_type(8)))  _Float16 v8h;
typedef __attribute__((ext_vector_type(8)))  float    v8f;
typedef __attribute__((ext_vector_type(4)))  int      v4i;

union HF16 { v16h v; v8h h[2]; };

// ---- CDNA5 async global->LDS path (ASYNCcnt), with portable fallback ----
#if __has_builtin(__builtin_amdgcn_global_load_async_to_lds_b128) && \
    __has_builtin(__builtin_amdgcn_s_wait_asynccnt)
#define HAS_ASYNC_LDS 1
#define WAIT_ASYNC(n) __builtin_amdgcn_s_wait_asynccnt(n)
#else
#define WAIT_ASYNC(n) do {} while (0)
#endif

// Builtin signature (from hipcc diagnostic): (v4i __device__*, v4i __shared__*,
// imm offset, imm cpol) -> pointee type is 4 x i32 (16 bytes).
typedef __attribute__((address_space(1))) v4i* as1_v4i_p;
typedef __attribute__((address_space(3))) v4i* as3_v4i_p;

// Copy 16 bytes global -> LDS (per active lane).
__device__ __forceinline__ void copy16_g2l(const _Float16* g, _Float16* l) {
#ifdef HAS_ASYNC_LDS
    __builtin_amdgcn_global_load_async_to_lds_b128(
        (as1_v4i_p)(g), (as3_v4i_p)(l), /*offset=*/0, /*cpol=*/0);
#else
    *(v8h*)l = *(const v8h*)g;
#endif
}

__device__ __forceinline__ v8f wmma_f16(v16h a, v16h b, v8f c) {
    return __builtin_amdgcn_wmma_f32_16x16x32_f16(
        /*neg_a=*/false, a, /*neg_b=*/false, b,
        /*c_mod=*/(short)0, c, /*reuse_a=*/false, /*reuse_b=*/false);
}

// Load a 16x32 A-style fragment from a row-major tile:
// lane (half = lane>>4, m16 = lane&15) reads row m16,
// K chunks [half*8, +8) and [16 + half*8, +8) -> two 16B loads.
__device__ __forceinline__ v16h load_frag_s(const _Float16* rowptr, int half) {
    HF16 u;
    u.h[0] = *(const v8h*)(rowptr + half * 8);
    u.h[1] = *(const v8h*)(rowptr + half * 8 + 16);
    return u.v;
}

// ---------------------------------------------------------------- convert
__global__ void cvt_f32_f16(const float* __restrict__ in,
                            _Float16* __restrict__ out, int n) {
    int i = blockIdx.x * blockDim.x + threadIdx.x;
    if (i < n) out[i] = (_Float16)in[i];
}

// ---------------------------------------------------------------- RoPE
// t layout: (B, S, nH, 64) f16; pair (2i, 2i+1) rotated by freqs[s][i].
__global__ void rope_kernel(_Float16* __restrict__ t,
                            const float* __restrict__ cs,
                            const float* __restrict__ sn,
                            int nH, int total) {
    int idx = blockIdx.x * blockDim.x + threadIdx.x;
    if (idx >= total) return;
    int i    = idx & 31;
    int rest = idx >> 5;                    // (b*S + s)*nH + h
    int s    = (rest / nH) & (S_LEN - 1);
    float c  = cs[s * 32 + i];
    float si = sn[s * 32 + i];
    _Float16* p = t + (size_t)rest * 64 + 2 * i;
    float tr = (float)p[0], ti = (float)p[1];
    p[0] = (_Float16)(tr * c - ti * si);
    p[1] = (_Float16)(tr * si + ti * c);
}

// ---------------------------------------------------------------- GEMM
// C[M x N] = A[M x K] * W[N x K]^T.
// Block = 256 thr = 8 waves (2M x 4N), block tile 128 x 256, wave tile 64x64.
// K-step 32, double-buffered async global->LDS staging: tiles are fetched
// once per workgroup into LDS (rows padded to 40 halfs), next K-step's tiles
// are in flight (ASYNCcnt) while the current one feeds 16 WMMAs per wave.
// STORE: 0 = f16 row-major, 1 = f16 transposed C^T[N][M], 2 = f32 row-major.
#define APAD 40   // row stride in halfs (32 data + 8 pad)

template <int STORE>
__global__ __launch_bounds__(256)
void gemm_nt(const _Float16* __restrict__ A, const _Float16* __restrict__ W,
             void* __restrict__ Cv, int K, int ldc) {
    __shared__ _Float16 sA[2][128 * APAD];   // 2 x 10 KB
    __shared__ _Float16 sB[2][256 * APAD];   // 2 x 20 KB

    const int tid  = threadIdx.x;
    const int lane = tid & 31;
    const int wave = tid >> 5;
    const int half = lane >> 4;
    const int m16  = lane & 15;
    const int wm   = wave >> 2;       // 0..1
    const int wn   = wave & 3;        // 0..3
    const int m0b  = blockIdx.x * 128;
    const int n0b  = blockIdx.y * 256;

    // Stage one K-step (32 cols) of A (128 rows) and B/W (256 rows) into LDS.
    // Per thread: 2 A-chunks + 4 B-chunks of 16B = 6 async ops in flight.
    auto stage = [&](int buf, int kk) {
#pragma unroll
        for (int j = 0; j < 2; j++) {
            int id = tid + 256 * j;         // 512 transfers
            int row = id >> 2, ch = id & 3;
            copy16_g2l(A + (size_t)(m0b + row) * K + kk + ch * 8,
                       &sA[buf][row * APAD + ch * 8]);
        }
#pragma unroll
        for (int j = 0; j < 4; j++) {
            int id = tid + 256 * j;         // 1024 transfers
            int row = id >> 2, ch = id & 3;
            copy16_g2l(W + (size_t)(n0b + row) * K + kk + ch * 8,
                       &sB[buf][row * APAD + ch * 8]);
        }
    };

    v8f c[4][4];
    const v8f zf = {0.f, 0.f, 0.f, 0.f, 0.f, 0.f, 0.f, 0.f};
#pragma unroll
    for (int i = 0; i < 4; i++)
#pragma unroll
        for (int j = 0; j < 4; j++) c[i][j] = zf;

    stage(0, 0);
    int buf = 0;
    for (int kk = 0; kk < K; kk += 32) {
        if (kk + 32 < K) {
            stage(buf ^ 1, kk + 32);    // prefetch next K-step
            WAIT_ASYNC(6);              // keep only the 6 newest in flight
        } else {
            WAIT_ASYNC(0);
        }
        __syncthreads();                // all waves' tiles visible

        v16h af[4], bf[4];
#pragma unroll
        for (int i = 0; i < 4; i++)
            af[i] = load_frag_s(&sA[buf][(wm * 64 + 16 * i + m16) * APAD],
                                half);
#pragma unroll
        for (int j = 0; j < 4; j++)
            bf[j] = load_frag_s(&sB[buf][(wn * 64 + 16 * j + m16) * APAD],
                                half);
#pragma unroll
        for (int i = 0; i < 4; i++)
#pragma unroll
            for (int j = 0; j < 4; j++)
                c[i][j] = wmma_f16(af[i], bf[j], c[i][j]);

        __syncthreads();                // done reading before overwrite
        buf ^= 1;
    }

#pragma unroll
    for (int i = 0; i < 4; i++)
#pragma unroll
        for (int j = 0; j < 4; j++)
#pragma unroll
            for (int r = 0; r < 8; r++) {
                int row = m0b + wm * 64 + 16 * i + r + 8 * half;
                int col = n0b + wn * 64 + 16 * j + m16;
                float v = c[i][j][r];
                if (STORE == 0)
                    ((_Float16*)Cv)[(size_t)row * ldc + col] = (_Float16)v;
                else if (STORE == 1)
                    ((_Float16*)Cv)[(size_t)col * ldc + row] = (_Float16)v;
                else
                    ((float*)Cv)[(size_t)row * ldc + col] = v;
            }
}

// ---------------------------------------------------------------- flash attn
// One wave per (b, h, 16-row q tile). Keys processed in blocks of 32.
// Q: (B,S,NH,64) f16 (post-RoPE), Km: (B,S,NKV,64) f16 (post-RoPE),
// Vt: (NKV*64, B*S) f16 (transposed V), Oa: (B,S,NH,64) f16.
__global__ __launch_bounds__(32)
void flash_attn(const _Float16* __restrict__ Q, const _Float16* __restrict__ Km,
                const _Float16* __restrict__ Vt, _Float16* __restrict__ Oa) {
    __shared__ _Float16 pbuf[16 * 32];   // per-wave P tile C-layout -> A-layout
    const int lane = threadIdx.x & 31;
    const int half = lane >> 4;
    const int m16  = lane & 15;
    const int bid  = blockIdx.x;
    const int qt   = bid & 127;          // S/16 = 128
    const int h    = (bid >> 7) & 31;
    const int b    = bid >> 12;
    const int kvh  = h >> 2;             // GQA group of 4
    const int q0   = qt * 16;
    const float scale = 0.125f;          // 1/sqrt(64)

    // Q fragments (16 rows x 64 d = two 16x32 A fragments)
    const _Float16* qrow =
        Q + ((size_t)((b * S_LEN + q0 + m16) * NH + h)) * HD;
    v16h aq0 = load_frag_s(qrow, half);
    v16h aq1 = load_frag_s(qrow + 32, half);

    const v8f zf = {0.f, 0.f, 0.f, 0.f, 0.f, 0.f, 0.f, 0.f};
    v8f o[4];
    float mrow[8], lrow[8];
#pragma unroll
    for (int t = 0; t < 4; t++) o[t] = zf;
#pragma unroll
    for (int r = 0; r < 8; r++) { mrow[r] = -1e30f; lrow[r] = 0.f; }

    const int jmax = q0 + 16;
    for (int j0 = 0; j0 < jmax; j0 += 32) {
        // ---- scores: two 16-key n-tiles, K-dim 64 split into two WMMAs
        float s[2][8];
#pragma unroll
        for (int nt = 0; nt < 2; nt++) {
            int key = j0 + nt * 16 + m16;
            const _Float16* krow =
                Km + ((size_t)((b * S_LEN + key) * NKV + kvh)) * HD;
            v16h bk0 = load_frag_s(krow, half);
            v16h bk1 = load_frag_s(krow + 32, half);
            v8f acc = zf;
            acc = wmma_f16(aq0, bk0, acc);
            acc = wmma_f16(aq1, bk1, acc);
#pragma unroll
            for (int r = 0; r < 8; r++) {
                int qr = q0 + r + 8 * half;
                float sv = acc[r] * scale;
                if (key > qr) sv = -1e30f;   // causal mask
                s[nt][r] = sv;
            }
        }
        // ---- online softmax: row stats across the 16 lanes of each half
        float mx[8];
#pragma unroll
        for (int r = 0; r < 8; r++) mx[r] = fmaxf(s[0][r], s[1][r]);
#pragma unroll
        for (int off = 8; off >= 1; off >>= 1)
#pragma unroll
            for (int r = 0; r < 8; r++)
                mx[r] = fmaxf(mx[r], __shfl_xor(mx[r], off, 32));

        float sc[8], rs[8];
#pragma unroll
        for (int r = 0; r < 8; r++) {
            float mn = fmaxf(mrow[r], mx[r]);
            sc[r]    = __expf(mrow[r] - mn);
            mrow[r]  = mn;
            rs[r]    = 0.f;
        }
#pragma unroll
        for (int nt = 0; nt < 2; nt++)
#pragma unroll
            for (int r = 0; r < 8; r++) {
                float p = __expf(s[nt][r] - mrow[r]);
                rs[r] += p;
                pbuf[(r + 8 * half) * 32 + nt * 16 + m16] = (_Float16)p;
            }
#pragma unroll
        for (int off = 8; off >= 1; off >>= 1)
#pragma unroll
            for (int r = 0; r < 8; r++)
                rs[r] += __shfl_xor(rs[r], off, 32);
#pragma unroll
        for (int r = 0; r < 8; r++) lrow[r] = lrow[r] * sc[r] + rs[r];
#pragma unroll
        for (int t = 0; t < 4; t++)
#pragma unroll
            for (int r = 0; r < 8; r++) o[t][r] *= sc[r];

        __syncthreads();  // DS stores visible before ds_load_b128 A-fragment
        // P as A-fragment (16x32) from LDS
        HF16 up;
        up.h[0] = *(const v8h*)&pbuf[m16 * 32 + half * 8];
        up.h[1] = *(const v8h*)&pbuf[m16 * 32 + half * 8 + 16];

        // ---- P (16x32) x V (32keys x 64d): 4 WMMAs, Vt rows contiguous in s
#pragma unroll
        for (int t = 0; t < 4; t++) {
            const _Float16* vrow =
                Vt + (size_t)(kvh * 64 + t * 16 + m16) * (size_t)MTOT +
                b * S_LEN + j0;
            v16h bv = load_frag_s(vrow, half);
            o[t] = wmma_f16(up.v, bv, o[t]);
        }
        __syncthreads();  // protect pbuf before next iteration's stores
    }

    // epilogue: normalize and write attn output (B,S,NH,64)
#pragma unroll
    for (int t = 0; t < 4; t++)
#pragma unroll
        for (int r = 0; r < 8; r++) {
            int row = q0 + r + 8 * half;
            int col = t * 16 + m16;
            float v = o[t][r] / lrow[r];
            Oa[((size_t)((b * S_LEN + row) * NH + h)) * HD + col] =
                (_Float16)v;
        }
}

// ---------------------------------------------------------------- launch
extern "C" void kernel_launch(void* const* d_in, const int* in_sizes, int n_in,
                              void* d_out, int out_size, void* d_ws,
                              size_t ws_size, hipStream_t stream) {
    const float* x  = (const float*)d_in[0];
    const float* fc = (const float*)d_in[1];
    const float* fs = (const float*)d_in[2];
    const float* wq = (const float*)d_in[3];
    const float* wk = (const float*)d_in[4];
    const float* wv = (const float*)d_in[5];
    const float* wo = (const float*)d_in[6];
    float* out = (float*)d_out;

    char* ws = (char*)d_ws;
    size_t off = 0;
    auto take = [&](size_t elems) {
        void* p = ws + off;
        off += (elems * sizeof(_Float16) + 255) & ~(size_t)255;
        return (_Float16*)p;
    };
    _Float16* x_h  = take((size_t)MTOT * DIMV);       // 16 MB
    _Float16* wq_h = take((size_t)DIMV * DIMV);       //  8 MB
    _Float16* wk_h = take((size_t)NKV * HD * DIMV);   //  2 MB
    _Float16* wv_h = take((size_t)NKV * HD * DIMV);   //  2 MB
    _Float16* wo_h = take((size_t)DIMV * DIMV);       //  8 MB
    _Float16* Qb   = take((size_t)MTOT * NH * HD);    // 16 MB
    _Float16* Kb   = take((size_t)MTOT * NKV * HD);   //  4 MB
    _Float16* Vtb  = take((size_t)NKV * HD * MTOT);   //  4 MB (transposed V)
    _Float16* Ab   = take((size_t)MTOT * NH * HD);    // 16 MB attn output

    auto cvt = [&](const float* src, _Float16* dst, int n) {
        cvt_f32_f16<<<(n + 255) / 256, 256, 0, stream>>>(src, dst, n);
    };
    cvt(x,  x_h,  MTOT * DIMV);
    cvt(wq, wq_h, DIMV * DIMV);
    cvt(wk, wk_h, NKV * HD * DIMV);
    cvt(wv, wv_h, NKV * HD * DIMV);
    cvt(wo, wo_h, DIMV * DIMV);

    // Projections: Q (N=2048), K (N=512), V transposed (N=512)
    gemm_nt<0><<<dim3(MTOT / 128, DIMV / 256), 256, 0, stream>>>(
        x_h, wq_h, Qb, DIMV, NH * HD);
    gemm_nt<0><<<dim3(MTOT / 128, (NKV * HD) / 256), 256, 0, stream>>>(
        x_h, wk_h, Kb, DIMV, NKV * HD);
    gemm_nt<1><<<dim3(MTOT / 128, (NKV * HD) / 256), 256, 0, stream>>>(
        x_h, wv_h, Vtb, DIMV, MTOT);

    // RoPE on Q and K
    {
        int totq = MTOT * NH * 32;
        rope_kernel<<<(totq + 255) / 256, 256, 0, stream>>>(Qb, fc, fs, NH,
                                                            totq);
        int totk = MTOT * NKV * 32;
        rope_kernel<<<(totk + 255) / 256, 256, 0, stream>>>(Kb, fc, fs, NKV,
                                                            totk);
    }

    // Flash attention: B * NH * (S/16) waves
    flash_attn<<<2 * NH * (S_LEN / 16), 32, 0, stream>>>(Qb, Kb, Vtb, Ab);

    // Output projection -> fp32 d_out
    gemm_nt<2><<<dim3(MTOT / 128, DIMV / 256), 256, 0, stream>>>(
        Ab, wo_h, out, NH * HD, DIMV);
}
